// LSTMCell_49967649521796
// MI455X (gfx1250) — compile-verified, matrix operand order checked by reading, not run on hardware
//
#include <hip/hip_runtime.h>
#include <stdint.h>

// ---------------------------------------------------------------------------
// Fused LSTM cell for MI455X (gfx1250, wave32, WMMA, async global->LDS).
//   gates = x @ Wx^T + h_prev @ Wh^T + b   (K fused to D+H = 2048)
//   c_t = sigmoid(f)*c_prev + sigmoid(i)*tanh(g);  h_t = sigmoid(o)*tanh(c_t)
//
// Pipeline:
//   1) cvt_xh / cvt_w : one-shot fp32 -> f16 pre-pass into d_ws
//        xh16 [8192 x 2048] = [x | h_prev] rows,  w16 [4096 x 2048] = [Wx | Wh]
//   2) lstm_gemm_async: double-buffered GLOBAL_LOAD_ASYNC_TO_LDS_B128 staging
//        (ASYNCcnt), v_wmma_f32_16x16x32_f16 with B-fragment lookahead,
//        fused activation epilogue (no gates tensor ever hits HBM).
// ---------------------------------------------------------------------------

typedef __attribute__((ext_vector_type(16))) _Float16 v16h;
typedef __attribute__((ext_vector_type(8)))  _Float16 h8;
typedef __attribute__((ext_vector_type(4)))  _Float16 h4;
typedef __attribute__((ext_vector_type(8)))  float    v8f;

#define B_DIM 8192
#define D_DIM 1024
#define H_DIM 1024
#define K_DIM (D_DIM + D_DIM)      // 2048

#define BM 128                      // block M tile
#define BN 64                       // block N tile per gate
#define KT 32                       // K per pipeline stage (one WMMA k-step)
#define STRIDE 40                   // halfs per LDS row: 32 + 8 pad = 80 B
#define A_BUF_BYTES (BM * STRIDE * 2)        // 10240
#define W_BUF_BYTES (4 * BN * STRIDE * 2)    // 20480
#define NSTAGE (K_DIM / KT)                  // 64

__device__ __forceinline__ float sigmoidf_(float x) {
    return 1.0f / (1.0f + __expf(-x));
}

// Branch-free tanh: 1 - 2/(e^{2x}+1).  Saturates correctly at +/-1.
__device__ __forceinline__ float tanhf_(float x) {
    const float e = __expf(2.0f * x);
    return 1.0f - 2.0f / (e + 1.0f);
}

__device__ __forceinline__ v16h combine16(h8 lo, h8 hi) {
    return __builtin_shufflevector(lo, hi, 0, 1, 2, 3, 4, 5, 6, 7,
                                           8, 9, 10, 11, 12, 13, 14, 15);
}

// Async 16-byte global -> LDS copy (ASYNCcnt). vdst = wave-relative LDS byte
// address, vaddr = 64-bit global address.  (CDNA5 ISA)
__device__ __forceinline__ void async_copy_b128(uint32_t lds_addr, const void* gptr) {
    asm volatile("global_load_async_to_lds_b128 %0, %1, off"
                 :
                 : "v"(lds_addr), "v"(gptr)
                 : "memory");
}

__device__ __forceinline__ void wait_asynccnt0() {
    asm volatile("s_wait_asynccnt 0x0" ::: "memory");
}

// ---------------------------------------------------------------------------
// Pre-pass: fp32 -> f16 conversion into workspace.
// ---------------------------------------------------------------------------
__global__ __launch_bounds__(256)
void cvt_xh(const float* __restrict__ x, const float* __restrict__ h,
            _Float16* __restrict__ out)
{
    const size_t t   = (size_t)blockIdx.x * 256 + threadIdx.x;   // B*K/8 threads
    const int    row = (int)(t >> 8);            // 256 8-elem chunks per row
    const int    c8  = ((int)t & 255) * 8;
    const float* src = (c8 < D_DIM) ? (x + (size_t)row * D_DIM + c8)
                                    : (h + (size_t)row * H_DIM + (c8 - D_DIM));
    const float4 f0 = ((const float4*)src)[0];
    const float4 f1 = ((const float4*)src)[1];
    h8 v = {(_Float16)f0.x, (_Float16)f0.y, (_Float16)f0.z, (_Float16)f0.w,
            (_Float16)f1.x, (_Float16)f1.y, (_Float16)f1.z, (_Float16)f1.w};
    *(h8*)(out + (size_t)row * K_DIM + c8) = v;
}

__global__ __launch_bounds__(256)
void cvt_w(const float* __restrict__ Wx, const float* __restrict__ Wh,
           _Float16* __restrict__ out)
{
    const size_t t   = (size_t)blockIdx.x * 256 + threadIdx.x;   // 4H*K/8 threads
    const int    row = (int)(t >> 8);
    const int    c8  = ((int)t & 255) * 8;
    const float* src = (c8 < D_DIM) ? (Wx + (size_t)row * D_DIM + c8)
                                    : (Wh + (size_t)row * H_DIM + (c8 - D_DIM));
    const float4 f0 = ((const float4*)src)[0];
    const float4 f1 = ((const float4*)src)[1];
    h8 v = {(_Float16)f0.x, (_Float16)f0.y, (_Float16)f0.z, (_Float16)f0.w,
            (_Float16)f1.x, (_Float16)f1.y, (_Float16)f1.z, (_Float16)f1.w};
    *(h8*)(out + (size_t)row * K_DIM + c8) = v;
}

// ---------------------------------------------------------------------------
// Main GEMM + LSTM epilogue, async double-buffered staging.
// ---------------------------------------------------------------------------
__global__ __launch_bounds__(256, 1)
void lstm_gemm_async(const _Float16* __restrict__ xh16,
                     const _Float16* __restrict__ w16,
                     const float* __restrict__ c_prev,
                     const float* __restrict__ bias,
                     float* __restrict__ h_out,
                     float* __restrict__ c_out)
{
    __shared__ _Float16 Alds[2][BM][STRIDE];        // 20 KB
    __shared__ _Float16 Wlds[2][4 * BN][STRIDE];    // 40 KB

    const int tid  = threadIdx.x;
    const int lane = tid & 31;
    const int wid  = tid >> 5;
    const int wm   = (wid >> 1) * 32;
    const int wn   = (wid & 1) * 32;

    const int m0 = blockIdx.y * BM;
    const int n0 = blockIdx.x * BN;

    const int lrow  = lane & 15;
    const int khalf = (lane >> 4) * 8;   // wave32 half-lane K offset (ISA layout)

    const uint32_t ldsA = (uint32_t)(uintptr_t)&Alds[0][0][0];
    const uint32_t ldsW = (uint32_t)(uintptr_t)&Wlds[0][0][0];

    v8f acc[4][2][2];
#pragma unroll
    for (int g = 0; g < 4; ++g)
#pragma unroll
        for (int i = 0; i < 2; ++i)
#pragma unroll
            for (int j = 0; j < 2; ++j)
                acc[g][i][j] = (v8f){0.f, 0.f, 0.f, 0.f, 0.f, 0.f, 0.f, 0.f};

    // Issue async copies for one KT-wide stage into buffer `buf`.
    auto issue_stage = [&](int buf, int kt) {
        // A tile: 128 rows x 64 B  -> 512 x 16 B chunks, 2 per thread.
#pragma unroll
        for (int ci = tid; ci < BM * 4; ci += 256) {
            const int r = ci >> 2;
            const int c = (ci & 3) * 8;                       // halfs
            const uint32_t dst = ldsA + (uint32_t)(buf * A_BUF_BYTES
                                 + r * (STRIDE * 2) + c * 2);
            const _Float16* src = xh16 + (size_t)(m0 + r) * K_DIM + kt + c;
            async_copy_b128(dst, src);
        }
        // W tiles (4 gates): 256 rows x 64 B -> 1024 chunks, 4 per thread.
#pragma unroll
        for (int ci = tid; ci < 4 * BN * 4; ci += 256) {
            const int r    = ci >> 2;                          // 0..255
            const int c    = (ci & 3) * 8;
            const int gate = r >> 6;
            const int nloc = r & 63;
            const uint32_t dst = ldsW + (uint32_t)(buf * W_BUF_BYTES
                                 + r * (STRIDE * 2) + c * 2);
            const _Float16* src = w16
                + (size_t)(gate * H_DIM + n0 + nloc) * K_DIM + kt + c;
            async_copy_b128(dst, src);
        }
    };

    // Prologue: fill buffer 0.
    issue_stage(0, 0);
    wait_asynccnt0();
    __syncthreads();

    for (int t = 0; t < NSTAGE; ++t) {
        const int cur = t & 1;
        if (t + 1 < NSTAGE)
            issue_stage(cur ^ 1, (t + 1) * KT);

        // ---- one WMMA k-step of 32 on the resident buffer ----
        const _Float16 (*Al)[STRIDE] = Alds[cur];
        const _Float16 (*Wl)[STRIDE] = Wlds[cur];

        v16h a[2];
#pragma unroll
        for (int i = 0; i < 2; ++i) {
            const int ar = wm + i * 16 + lrow;
            a[i] = combine16(*(const h8*)&Al[ar][khalf],
                             *(const h8*)&Al[ar][khalf + 16]);
        }

        // B-fragment lookahead: load gate g+1 while gate g's WMMAs execute,
        // so s_wait_dscnt retires behind 4 in-flight WMMAs.
        v16h bcur[2], bnext[2];
#pragma unroll
        for (int j = 0; j < 2; ++j) {
            const int br = wn + j * 16 + lrow;                 // gate 0
            bcur[j] = combine16(*(const h8*)&Wl[br][khalf],
                                *(const h8*)&Wl[br][khalf + 16]);
        }
#pragma unroll
        for (int g = 0; g < 4; ++g) {
            if (g < 3) {
#pragma unroll
                for (int j = 0; j < 2; ++j) {
                    const int br = (g + 1) * BN + wn + j * 16 + lrow;
                    bnext[j] = combine16(*(const h8*)&Wl[br][khalf],
                                         *(const h8*)&Wl[br][khalf + 16]);
                }
            }
#pragma unroll
            for (int j = 0; j < 2; ++j)
#pragma unroll
                for (int i = 0; i < 2; ++i)
                    acc[g][i][j] = __builtin_amdgcn_wmma_f32_16x16x32_f16(
                        false, a[i], false, bcur[j],
                        (short)0, acc[g][i][j], false, false);
#pragma unroll
            for (int j = 0; j < 2; ++j)
                bcur[j] = bnext[j];
        }

        if (t + 1 < NSTAGE) {
            wait_asynccnt0();   // next buffer fully in LDS (this wave's copies)
            __syncthreads();    // ...and everyone else's; cur buffer drained
        }
    }

    // ---- fused LSTM epilogue ----
#pragma unroll
    for (int j = 0; j < 2; ++j) {
        const int n  = n0 + wn + j * 16 + lrow;
        const float bi = bias[n];
        const float bf = bias[H_DIM + n];
        const float bo = bias[2 * H_DIM + n];
        const float bg = bias[3 * H_DIM + n];
#pragma unroll
        for (int i = 0; i < 2; ++i) {
            const int mbase = m0 + wm + i * 16 + (lane >> 4) * 8;
#pragma unroll
            for (int v = 0; v < 8; ++v) {
                const int m = mbase + v;
                const float gi = sigmoidf_(acc[0][i][j][v] + bi);
                const float gf = sigmoidf_(acc[1][i][j][v] + bf);
                const float go = sigmoidf_(acc[2][i][j][v] + bo);
                const float gg = tanhf_(acc[3][i][j][v] + bg);
                const size_t idx = (size_t)m * H_DIM + n;
                const float c = gf * c_prev[idx] + gi * gg;
                h_out[idx] = go * tanhf_(c);
                c_out[idx] = c;
            }
        }
    }
}

// ---------------------------------------------------------------------------
// Fallback (no-workspace path): used only if d_ws is too small.
// ---------------------------------------------------------------------------
__global__ __launch_bounds__(256, 1)
void lstm_fused_fallback(const float* __restrict__ x,
                         const float* __restrict__ h_prev,
                         const float* __restrict__ c_prev,
                         const float* __restrict__ Wx,
                         const float* __restrict__ Wh,
                         const float* __restrict__ bias,
                         float* __restrict__ h_out,
                         float* __restrict__ c_out)
{
    __shared__ _Float16 Alds[BM][72];
    __shared__ _Float16 Wlds[4 * BN][72];

    const int tid  = threadIdx.x;
    const int lane = tid & 31;
    const int wid  = tid >> 5;
    const int wm   = (wid >> 1) * 32;
    const int wn   = (wid & 1) * 32;
    const int m0   = blockIdx.y * BM;
    const int n0   = blockIdx.x * BN;
    const int lrow  = lane & 15;
    const int khalf = (lane >> 4) * 8;

    v8f acc[4][2][2];
#pragma unroll
    for (int g = 0; g < 4; ++g)
#pragma unroll
        for (int i = 0; i < 2; ++i)
#pragma unroll
            for (int j = 0; j < 2; ++j)
                acc[g][i][j] = (v8f){0.f, 0.f, 0.f, 0.f, 0.f, 0.f, 0.f, 0.f};

    for (int kt = 0; kt < K_DIM; kt += 64) {
        const float* Abase = (kt < D_DIM) ? (x + (size_t)kt)
                                          : (h_prev + (size_t)(kt - D_DIM));
#pragma unroll
        for (int ci = tid; ci < BM * 16; ci += 256) {
            const int r  = ci >> 4;
            const int c4 = (ci & 15) * 4;
            const float4 f = *(const float4*)(Abase + (size_t)(m0 + r) * 1024 + c4);
            h4 v = {(_Float16)f.x, (_Float16)f.y, (_Float16)f.z, (_Float16)f.w};
            *(h4*)&Alds[r][c4] = v;
        }
        const float* Wbase = (kt < D_DIM) ? (Wx + (size_t)kt)
                                          : (Wh + (size_t)(kt - D_DIM));
#pragma unroll
        for (int ci = tid; ci < 4 * BN * 16; ci += 256) {
            const int r    = ci >> 4;
            const int c4   = (ci & 15) * 4;
            const size_t grow = (size_t)((r >> 6) * H_DIM + n0 + (r & 63));
            const float4 f = *(const float4*)(Wbase + grow * 1024 + c4);
            h4 v = {(_Float16)f.x, (_Float16)f.y, (_Float16)f.z, (_Float16)f.w};
            *(h4*)&Wlds[r][c4] = v;
        }
        __syncthreads();
#pragma unroll
        for (int ks = 0; ks < 64; ks += 32) {
            v16h a[2];
#pragma unroll
            for (int i = 0; i < 2; ++i) {
                const int ar = wm + i * 16 + lrow;
                a[i] = combine16(*(const h8*)&Alds[ar][ks + khalf],
                                 *(const h8*)&Alds[ar][ks + khalf + 16]);
            }
#pragma unroll
            for (int g = 0; g < 4; ++g)
#pragma unroll
                for (int j = 0; j < 2; ++j) {
                    const int br = g * BN + wn + j * 16 + lrow;
                    const v16h bf = combine16(*(const h8*)&Wlds[br][ks + khalf],
                                              *(const h8*)&Wlds[br][ks + khalf + 16]);
#pragma unroll
                    for (int i = 0; i < 2; ++i)
                        acc[g][i][j] = __builtin_amdgcn_wmma_f32_16x16x32_f16(
                            false, a[i], false, bf, (short)0, acc[g][i][j],
                            false, false);
                }
        }
        __syncthreads();
    }

#pragma unroll
    for (int j = 0; j < 2; ++j) {
        const int n  = n0 + wn + j * 16 + lrow;
        const float bi = bias[n];
        const float bf = bias[H_DIM + n];
        const float bo = bias[2 * H_DIM + n];
        const float bg = bias[3 * H_DIM + n];
#pragma unroll
        for (int i = 0; i < 2; ++i) {
            const int mbase = m0 + wm + i * 16 + (lane >> 4) * 8;
#pragma unroll
            for (int v = 0; v < 8; ++v) {
                const int m = mbase + v;
                const float gi = sigmoidf_(acc[0][i][j][v] + bi);
                const float gf = sigmoidf_(acc[1][i][j][v] + bf);
                const float go = sigmoidf_(acc[2][i][j][v] + bo);
                const float gg = tanhf_(acc[3][i][j][v] + bg);
                const size_t idx = (size_t)m * H_DIM + n;
                const float c = gf * c_prev[idx] + gi * gg;
                h_out[idx] = go * tanhf_(c);
                c_out[idx] = c;
            }
        }
    }
}

extern "C" void kernel_launch(void* const* d_in, const int* in_sizes, int n_in,
                              void* d_out, int out_size, void* d_ws, size_t ws_size,
                              hipStream_t stream) {
    const float* x      = (const float*)d_in[0];
    const float* h_prev = (const float*)d_in[1];
    const float* c_prev = (const float*)d_in[2];
    const float* Wx     = (const float*)d_in[3];
    const float* Wh     = (const float*)d_in[4];
    const float* b      = (const float*)d_in[5];

    float* h_out = (float*)d_out;                     // h_t first
    float* c_out = h_out + (size_t)B_DIM * H_DIM;     // then c_t

    const size_t xh_elems = (size_t)B_DIM * K_DIM;        // 16.8M f16
    const size_t w_elems  = (size_t)(4 * H_DIM) * K_DIM;  // 8.4M f16
    const size_t need     = (xh_elems + w_elems) * sizeof(_Float16);  // 48 MB

    dim3 grid(H_DIM / BN, B_DIM / BM);   // (16, 64)

    if (ws_size >= need) {
        _Float16* xh16 = (_Float16*)d_ws;
        _Float16* w16  = xh16 + xh_elems;
        cvt_xh<<<(unsigned)(xh_elems / 8 / 256), 256, 0, stream>>>(x, h_prev, xh16);
        cvt_w <<<(unsigned)(w_elems / 8 / 256), 256, 0, stream>>>(Wx, Wh, w16);
        lstm_gemm_async<<<grid, 256, 0, stream>>>(xh16, w16, c_prev, b,
                                                  h_out, c_out);
    } else {
        lstm_fused_fallback<<<grid, 256, 0, stream>>>(x, h_prev, c_prev,
                                                      Wx, Wh, b, h_out, c_out);
    }
}
